// LearntNeighbourhoodSamplingV2_19198503813328
// MI455X (gfx1250) — matrix operand order repaired and checked by reading, not compiled, not run on hardware
//
#include <hip/hip_runtime.h>

typedef __attribute__((ext_vector_type(16))) _Float16 v16h;
typedef __attribute__((ext_vector_type(8)))  float    v8f;

#define B_   2
#define C_   128
#define H_   800
#define W_   640
#define N_   10000          // divisible by 16: 625 WMMA tiles per batch, no padding
#define O_   18             // conv output channels
#define OPAD 32             // conv rows padded to 32 (two 16-wide WMMA n-tiles)

// Bilinear setup, faithful to grid_sample(align_corners=True, padding_mode='border')
__device__ __forceinline__ void bilin_setup(float vx, float vy,
                                            int& x0, int& x1, int& y0, int& y1,
                                            float& w00, float& w01, float& w10, float& w11) {
  float gx = (vx + 1.0f) * 0.5f * (float)(W_ - 1);
  float gy = (vy + 1.0f) * 0.5f * (float)(H_ - 1);
  gx = fminf(fmaxf(gx, 0.0f), (float)(W_ - 1));
  gy = fminf(fmaxf(gy, 0.0f), (float)(H_ - 1));
  float x0f = floorf(gx), y0f = floorf(gy);
  float wx1 = gx - x0f,   wy1 = gy - y0f;
  x0 = (int)x0f;  y0 = (int)y0f;
  x1 = x0 + 1 < W_ - 1 ? x0 + 1 : W_ - 1;
  y1 = y0 + 1 < H_ - 1 ? y0 + 1 : H_ - 1;
  float wx0 = 1.0f - wx1, wy0 = 1.0f - wy1;
  w00 = wy0 * wx0; w01 = wy0 * wx1; w10 = wy1 * wx0; w11 = wy1 * wx1;
}

// ---------------- Stage 1: center gather -> feats (f16, [B][N][C]) ----------------
// One wave32 per point; each lane produces 4 channels (128 = 32 lanes * 4).
__global__ void k_center(const float* __restrict__ img, const float* __restrict__ verts,
                         _Float16* __restrict__ feats) {
  int wave = (blockIdx.x * blockDim.x + threadIdx.x) >> 5;
  int lane = threadIdx.x & 31;
  if (wave >= B_ * N_) return;                 // wave-uniform exit
  int b = wave / N_, n = wave % N_;

  float vx = verts[((long)b * N_ + n) * 2 + 0];
  float vy = verts[((long)b * N_ + n) * 2 + 1];
  int x0, x1, y0, y1; float w00, w01, w10, w11;
  bilin_setup(vx, vy, x0, x1, y0, y1, w00, w01, w10, w11);

  long o00 = (long)y0 * W_ + x0, o01 = (long)y0 * W_ + x1;
  long o10 = (long)y1 * W_ + x0, o11 = (long)y1 * W_ + x1;
  const float* ib = img + (long)b * C_ * H_ * W_;
  _Float16* frow = feats + ((long)b * N_ + n) * C_;

#pragma unroll
  for (int i = 0; i < 4; ++i) {
    int c = lane * 4 + i;
    const float* pc = ib + (long)c * (H_ * W_);
    float v = pc[o00] * w00 + pc[o01] * w01 + pc[o10] * w10 + pc[o11] * w11;
    frow[c] = (_Float16)v;
  }
}

// ---------------- Stage 2: delta = feats * conv_w^T + bias via WMMA ----------------
// One wave per tile of 16 points. M=16 points, K=128 channels (4 k-steps of 32),
// 18 outputs as two 16-wide n-tiles (second tile zero-padded via LDS staging).
__global__ void k_delta(const _Float16* __restrict__ feats, const float* __restrict__ Wc,
                        const float* __restrict__ bias, float* __restrict__ delta) {
  __shared__ _Float16 wsh[OPAD * C_];          // conv_w in f16, rows 18..31 zeroed (8 KB)
  __shared__ float    bsh[OPAD];               // bias, entries 18..31 zeroed

  // Cooperative staging: 256 threads x 16 halfs = 4096 = 32*128
  for (int i = 0; i < 16; ++i) {
    int idx = threadIdx.x * 16 + i;
    int row = idx >> 7, col = idx & 127;
    wsh[idx] = (row < O_) ? (_Float16)Wc[row * C_ + col] : (_Float16)0.0f;
  }
  if (threadIdx.x < OPAD) bsh[threadIdx.x] = (threadIdx.x < O_) ? bias[threadIdx.x] : 0.0f;
  __syncthreads();

  const int TPB = N_ / 16;                     // 625 tiles per batch
  int wave = (blockIdx.x * blockDim.x + threadIdx.x) >> 5;
  int lane = threadIdx.x & 31;
  if (wave >= B_ * TPB) return;                // wave-uniform exit: EXEC all-1s at WMMA
  int b = wave / TPB, tb = wave % TPB;

  int m   = lane & 15;                         // A row / D column index
  int kbA = (lane < 16) ? 0 : 8;               // A layout: low lanes K {0-7,16-23}, high {8-15,24-31}
  int kbB = (lane < 16) ? 0 : 8;               // B layout: low lanes K {0-7,16-23}, high {8-15,24-31}
  const _Float16* rowA = feats + ((long)b * N_ + (long)tb * 16 + m) * C_;

  // Build all tile-invariant B fragments from LDS (uniform control flow, no exec masking)
  v16h bf0[4], bf1[4];
#pragma unroll
  for (int ks = 0; ks < 4; ++ks) {
#pragma unroll
    for (int j = 0; j < 8; ++j) {
      int c = ks * 32 + ((j < 4) ? (kbB + 2 * j) : (16 + kbB + 2 * (j - 4)));
      bf0[ks][2 * j]     = wsh[m * C_ + c];
      bf0[ks][2 * j + 1] = wsh[m * C_ + c + 1];
      bf1[ks][2 * j]     = wsh[(16 + m) * C_ + c];
      bf1[ks][2 * j + 1] = wsh[(16 + m) * C_ + c + 1];
    }
  }

  v8f acc0 = {}, acc1 = {};
#pragma unroll
  for (int ks = 0; ks < 4; ++ks) {
    v16h a;
#pragma unroll
    for (int j = 0; j < 8; ++j) {              // ISA 16-bit A layout: VGPR j holds a K pair
      int kl = (j < 4) ? (kbA + 2 * j) : (16 + kbA + 2 * (j - 4));
      int c  = ks * 32 + kl;
      a[2 * j]     = rowA[c];
      a[2 * j + 1] = rowA[c + 1];
    }
    acc0 = __builtin_amdgcn_wmma_f32_16x16x32_f16(false, a, false, bf0[ks], (short)0, acc0, false, false);
    acc1 = __builtin_amdgcn_wmma_f32_16x16x32_f16(false, a, false, bf1[ks], (short)0, acc1, false, false);
  }

  // D layout: VGPR r holds point M = r (+8 for lanes 16-31), output index = lane&15
  int mo = (lane >= 16) ? 8 : 0;
#pragma unroll
  for (int r = 0; r < 8; ++r) {
    int npt = tb * 16 + r + mo;                // always < N_ (625*16 == 10000)
    float* drow = delta + ((long)b * N_ + npt) * O_;
    drow[m] = acc0[r] + bsh[m];
    if (m < O_ - 16) drow[16 + m] = acc1[r] + bsh[16 + m];
  }
}

// ---------------- Stage 3: 9-neighbor gather + mean -> out (B,N,C) ----------------
__global__ void k_neigh(const float* __restrict__ img, const float* __restrict__ verts,
                        const float* __restrict__ delta, float* __restrict__ out) {
  int wave = (blockIdx.x * blockDim.x + threadIdx.x) >> 5;
  int lane = threadIdx.x & 31;
  if (wave >= B_ * N_) return;
  int b = wave / N_, n = wave % N_;

  float vx = verts[((long)b * N_ + n) * 2 + 0];
  float vy = verts[((long)b * N_ + n) * 2 + 1];
  const float* drow = delta + ((long)b * N_ + n) * O_;
  const float* ib   = img + (long)b * C_ * H_ * W_;

  float acc[4] = {0.0f, 0.0f, 0.0f, 0.0f};
  for (int j = 0; j < 9; ++j) {
    float dx = (j == 0) ? 0.0f : drow[2 * j];      // delta[:,:,0,:] is zeroed in reference
    float dy = (j == 0) ? 0.0f : drow[2 * j + 1];
    int x0, x1, y0, y1; float w00, w01, w10, w11;
    bilin_setup(vx + dx, vy + dy, x0, x1, y0, y1, w00, w01, w10, w11);
    long o00 = (long)y0 * W_ + x0, o01 = (long)y0 * W_ + x1;
    long o10 = (long)y1 * W_ + x0, o11 = (long)y1 * W_ + x1;
#pragma unroll
    for (int i = 0; i < 4; ++i) {
      int c = lane * 4 + i;
      const float* pc = ib + (long)c * (H_ * W_);
      acc[i] += pc[o00] * w00 + pc[o01] * w01 + pc[o10] * w10 + pc[o11] * w11;
    }
  }
  float* orow = out + ((long)b * N_ + n) * C_;
#pragma unroll
  for (int i = 0; i < 4; ++i) orow[lane * 4 + i] = acc[i] * (1.0f / 9.0f);
}

extern "C" void kernel_launch(void* const* d_in, const int* in_sizes, int n_in,
                              void* d_out, int out_size, void* d_ws, size_t ws_size,
                              hipStream_t stream) {
  const float* img  = (const float*)d_in[0];   // (2,128,800,640) f32
  const float* vrt  = (const float*)d_in[1];   // (2,10000,2)     f32
  const float* Wc   = (const float*)d_in[2];   // (18,128)        f32
  const float* bias = (const float*)d_in[3];   // (18,)           f32
  float* out = (float*)d_out;                  // (2,10000,128)   f32

  char* ws = (char*)d_ws;
  _Float16* feats = (_Float16*)ws;                                   // B*N*C f16  (~5.1 MB)
  size_t featsBytes = (size_t)B_ * N_ * C_ * sizeof(_Float16);
  float* delta = (float*)(ws + ((featsBytes + 255) & ~(size_t)255)); // B*N*O f32  (~1.4 MB)

  int wavesPts = B_ * N_;                       // one wave per point
  int blkPts   = (wavesPts * 32 + 255) / 256;
  int tiles    = B_ * (N_ / 16);                // one wave per 16-point tile
  int blkTiles = (tiles * 32 + 255) / 256;

  k_center<<<blkPts,   256, 0, stream>>>(img, vrt, feats);
  k_delta <<<blkTiles, 256, 0, stream>>>(feats, Wc, bias, delta);
  k_neigh <<<blkPts,   256, 0, stream>>>(img, vrt, delta, out);
}
